// SelfAttention_6081673691663
// MI455X (gfx1250) — matrix operand order repaired and checked by reading, not compiled, not run on hardware
//
#include <hip/hip_runtime.h>
#include <stdint.h>

typedef __attribute__((ext_vector_type(16))) _Float16 v16h;
typedef __attribute__((ext_vector_type(8)))  _Float16 v8h;
typedef __attribute__((ext_vector_type(4)))  _Float16 v4h;
typedef __attribute__((ext_vector_type(8)))  float    v8f;

union H16 { v16h v; v8h h[2]; };

constexpr int kB  = 4;
constexpr int kS  = 2048;
constexpr int kH  = 16;
constexpr int kD  = 128;
constexpr int kBM = 64;   // queries per workgroup (4 waves x 16 rows)
constexpr int kBN = 32;   // keys per inner tile
constexpr int KST = 136;  // kbuf row stride in halves (128 + 8 pad -> conflict-free)
constexpr int VST = 40;   // vtbuf row stride in halves (32 + 8 pad)
constexpr int PST = 40;   // pbuf row stride in halves (32 + 8 pad)
constexpr float kNeg   = -10000.0f;
constexpr float kLog2e = 1.4426950408889634f;

// Issue one 32x128 fp32 tile (K or V) as 8 async b128 DMAs per lane.
// offset: advances BOTH the global address and the LDS address (ISA 08 4.4).
__device__ __forceinline__ void async_tile_load(const float* gp, uint32_t lds) {
  const uint64_t ga = (uint64_t)(uintptr_t)gp;
  asm volatile("global_load_async_to_lds_b128 %0, %1, off offset:0"   :: "v"(lds), "v"(ga) : "memory");
  asm volatile("global_load_async_to_lds_b128 %0, %1, off offset:16"  :: "v"(lds), "v"(ga) : "memory");
  asm volatile("global_load_async_to_lds_b128 %0, %1, off offset:32"  :: "v"(lds), "v"(ga) : "memory");
  asm volatile("global_load_async_to_lds_b128 %0, %1, off offset:48"  :: "v"(lds), "v"(ga) : "memory");
  asm volatile("global_load_async_to_lds_b128 %0, %1, off offset:64"  :: "v"(lds), "v"(ga) : "memory");
  asm volatile("global_load_async_to_lds_b128 %0, %1, off offset:80"  :: "v"(lds), "v"(ga) : "memory");
  asm volatile("global_load_async_to_lds_b128 %0, %1, off offset:96"  :: "v"(lds), "v"(ga) : "memory");
  asm volatile("global_load_async_to_lds_b128 %0, %1, off offset:112" :: "v"(lds), "v"(ga) : "memory");
}

__global__ __launch_bounds__(128)
void fa_fwd_f16wmma(const float* __restrict__ qkv, float* __restrict__ out) {
  __shared__ float    kstage[2][kBN * kD];  // raw fp32 K tiles (async DMA dst)
  __shared__ float    vstage[2][kBN * kD];  // raw fp32 V tiles (async DMA dst)
  __shared__ _Float16 kbuf[kBN * KST];      // scaled f16 K tile, [key][dim]
  __shared__ _Float16 vtbuf[kD * VST];      // f16 V tile transposed, [dim][key]
  __shared__ _Float16 pbuf[4 * 16 * PST];   // per-wave P relayout scratch

  const int qb = blockIdx.x;   // query block 0..31
  const int h  = blockIdx.y;   // head 0..15
  const int b  = blockIdx.z;   // batch 0..3

  const int tid  = threadIdx.x;
  const int wv   = tid >> 5;
  const int lane = tid & 31;
  const int l16  = lane & 15;
  const int hf   = lane >> 4;  // 0: lanes 0-15, 1: lanes 16-31

  const size_t sstr = (size_t)3 * kH * kD;                    // 6144 floats per (b,s)
  const float* qbase = qkv + (size_t)b * kS * sstr + (size_t)h * kD;
  const float* kbase = qbase + (size_t)kH * kD;               // +2048
  const float* vbase = qbase + (size_t)2 * kH * kD;           // +4096

  const int qs0  = qb * kBM + wv * 16;  // first query row of this wave
  const int qrow = qs0 + l16;           // this lane's query row (A-frag row)
  const float scale = 0.088388347648318447f;  // 1/sqrt(128)

  // per-thread DMA / convert slice: one key row x 32 dims
  const int drow = tid >> 2;            // key 0..31 within tile
  const int dc0  = (tid & 3) * 32;      // dim offset 0/32/64/96

  // ---- Q fragments (16-bit A-matrix 16x32 layout), straight from global ----
  v16h qfrag[4];
  {
    const float* qp = qbase + (size_t)qrow * sstr;
    #pragma unroll
    for (int f = 0; f < 4; ++f) {
      const int k0 = 32 * f + 8 * hf;              // halves 0-7 <- K k0..k0+7
      float4 a0 = *(const float4*)(qp + k0);
      float4 a1 = *(const float4*)(qp + k0 + 4);
      float4 b0 = *(const float4*)(qp + k0 + 16);  // halves 8-15 <- K k0+16..+23
      float4 b1 = *(const float4*)(qp + k0 + 20);
      v16h q;
      q[0]  = (_Float16)a0.x; q[1]  = (_Float16)a0.y;
      q[2]  = (_Float16)a0.z; q[3]  = (_Float16)a0.w;
      q[4]  = (_Float16)a1.x; q[5]  = (_Float16)a1.y;
      q[6]  = (_Float16)a1.z; q[7]  = (_Float16)a1.w;
      q[8]  = (_Float16)b0.x; q[9]  = (_Float16)b0.y;
      q[10] = (_Float16)b0.z; q[11] = (_Float16)b0.w;
      q[12] = (_Float16)b1.x; q[13] = (_Float16)b1.y;
      q[14] = (_Float16)b1.z; q[15] = (_Float16)b1.w;
      qfrag[f] = q;
    }
  }

  v8f acc[8];                    // O accumulators, 8 tiles of 16 dims (C-layout)
  #pragma unroll
  for (int t = 0; t < 8; ++t) acc[t] = {};
  float m_i[8], l_i[8];          // per-row online-softmax state (rows r + 8*hf)
  #pragma unroll
  for (int r = 0; r < 8; ++r) { m_i[r] = -3.0e38f; l_i[r] = 0.0f; }

  _Float16* pw = &pbuf[wv * 16 * PST];
  const int nkb = 2 * qb + 2;    // causal: only key tiles covering [0, qb*64+64)

  // ---- prologue: DMA tile 0 into stage[0] ----
  {
    const uint32_t kl = (uint32_t)(uintptr_t)&kstage[0][drow * kD + dc0];
    const uint32_t vl = (uint32_t)(uintptr_t)&vstage[0][drow * kD + dc0];
    async_tile_load(kbase + (size_t)drow * sstr + dc0, kl);
    async_tile_load(vbase + (size_t)drow * sstr + dc0, vl);
  }

  for (int kb = 0; kb < nkb; ++kb) {
    const int ks = kb * kBN;
    const int p  = kb & 1;

    // ---- issue DMA for tile kb+1 into the other stage buffer ----
    if (kb + 1 < nkb) {
      const int ks1 = ks + kBN;
      const uint32_t kl = (uint32_t)(uintptr_t)&kstage[p ^ 1][drow * kD + dc0];
      const uint32_t vl = (uint32_t)(uintptr_t)&vstage[p ^ 1][drow * kD + dc0];
      async_tile_load(kbase + (size_t)(ks1 + drow) * sstr + dc0, kl);
      async_tile_load(vbase + (size_t)(ks1 + drow) * sstr + dc0, vl);
      // async ops complete in order: <=16 outstanding => tile kb's 16 are done
      asm volatile("s_wait_asynccnt 0x10" ::: "memory");
    } else {
      asm volatile("s_wait_asynccnt 0x0" ::: "memory");
    }
    __syncthreads();   // all waves' DMAs for tile kb landed; prev compute done

    // ---- cooperative convert: stage[p] fp32 -> f16 kbuf (scaled) / vtbuf^T ----
    {
      const float* ksrc = &kstage[p][drow * kD + dc0];
      const float* vsrc = &vstage[p][drow * kD + dc0];
      _Float16* kd = &kbuf[drow * KST + dc0];
      #pragma unroll
      for (int j = 0; j < 32; j += 4) {
        float4 kv = *(const float4*)(ksrc + j);
        v4h pk;                                   // packed -> ds_store_b64
        pk[0] = (_Float16)(kv.x * scale);
        pk[1] = (_Float16)(kv.y * scale);
        pk[2] = (_Float16)(kv.z * scale);
        pk[3] = (_Float16)(kv.w * scale);
        *(v4h*)(kd + j) = pk;
        float4 vv = *(const float4*)(vsrc + j);   // transpose scatter (b16 each)
        vtbuf[(dc0 + j + 0) * VST + drow] = (_Float16)vv.x;
        vtbuf[(dc0 + j + 1) * VST + drow] = (_Float16)vv.y;
        vtbuf[(dc0 + j + 2) * VST + drow] = (_Float16)vv.z;
        vtbuf[(dc0 + j + 3) * VST + drow] = (_Float16)vv.w;
      }
    }
    __syncthreads();

    // ---- per-wave causal skip: tile fully masked for this wave's 16 rows ----
    if (ks <= qs0 + 15) {
      // ---- S = Q * K^T : two 16-key tiles, K-dim 128 = 4 WMMAs each ----
      v8f st[2];
      #pragma unroll
      for (int n = 0; n < 2; ++n) {
        v8f c = {};
        #pragma unroll
        for (int f = 0; f < 4; ++f) {
          const _Float16* bp = &kbuf[(n * 16 + l16) * KST + 32 * f + 8 * hf];
          H16 bf;
          bf.h[0] = *(const v8h*)bp;
          bf.h[1] = *(const v8h*)(bp + 16);
          c = __builtin_amdgcn_wmma_f32_16x16x32_f16(false, qfrag[f], false, bf.v,
                                                     (short)0, c, false, false);
        }
        st[n] = c;
      }

      // ---- causal mask (reference semantics: add -10000 where s > t) ----
      if (ks + kBN - 1 > qs0) {            // wave-uniform condition
        #pragma unroll
        for (int n = 0; n < 2; ++n) {
          const int sidx = ks + n * 16 + l16;
          #pragma unroll
          for (int r = 0; r < 8; ++r) {
            const int tidx = qs0 + r + 8 * hf;
            if (sidx > tidx) st[n][r] += kNeg;
          }
        }
      }

      // ---- online softmax: rowwise max/sum via width-16 xor shuffles ----
      float alpha[8];
      #pragma unroll
      for (int r = 0; r < 8; ++r) {
        float x = fmaxf(st[0][r], st[1][r]);
        #pragma unroll
        for (int off = 1; off < 16; off <<= 1)
          x = fmaxf(x, __shfl_xor(x, off, 16));
        const float mn = fmaxf(m_i[r], x);
        alpha[r] = __builtin_amdgcn_exp2f((m_i[r] - mn) * kLog2e);
        const float p0 = __builtin_amdgcn_exp2f((st[0][r] - mn) * kLog2e);
        const float p1 = __builtin_amdgcn_exp2f((st[1][r] - mn) * kLog2e);
        st[0][r] = p0; st[1][r] = p1;
        float sum = p0 + p1;
        #pragma unroll
        for (int off = 1; off < 16; off <<= 1)
          sum += __shfl_xor(sum, off, 16);
        l_i[r] = l_i[r] * alpha[r] + sum;
        m_i[r] = mn;
      }
      #pragma unroll
      for (int t = 0; t < 8; ++t) {
        #pragma unroll
        for (int r = 0; r < 8; ++r) acc[t][r] *= alpha[r];
      }

      // ---- relayout P (C-layout f32 -> A-layout f16) via wave-private LDS ----
      #pragma unroll
      for (int r = 0; r < 8; ++r) {
        const int row = r + 8 * hf;
        pw[row * PST + l16]      = (_Float16)st[0][r];
        pw[row * PST + 16 + l16] = (_Float16)st[1][r];
      }
      // same-wave DS ops complete in order: safe to read back without barrier
      H16 pf;
      const _Float16* pr = &pw[l16 * PST + 8 * hf];
      pf.h[0] = *(const v8h*)pr;         // cols 8*hf .. +7
      pf.h[1] = *(const v8h*)(pr + 16);  // cols 16+8*hf .. +7

      // ---- O += P * V : 8 dim-tiles of 16 ----
      #pragma unroll
      for (int t = 0; t < 8; ++t) {
        const _Float16* vp2 = &vtbuf[(t * 16 + l16) * VST + 8 * hf];
        H16 vf;
        vf.h[0] = *(const v8h*)vp2;
        vf.h[1] = *(const v8h*)(vp2 + 16);
        acc[t] = __builtin_amdgcn_wmma_f32_16x16x32_f16(false, pf.v, false, vf.v,
                                                        (short)0, acc[t], false, false);
      }
    }
  }

  // ---- epilogue: O / l, fp32 store in bthd layout ----
  float inv[8];
  #pragma unroll
  for (int r = 0; r < 8; ++r) inv[r] = 1.0f / l_i[r];
  #pragma unroll
  for (int t = 0; t < 8; ++t) {
    #pragma unroll
    for (int r = 0; r < 8; ++r) {
      const int trow = qs0 + r + 8 * hf;
      out[((size_t)(b * kS + trow) * kH + h) * kD + t * 16 + l16] =
          acc[t][r] * inv[r];
    }
  }
}

extern "C" void kernel_launch(void* const* d_in, const int* in_sizes, int n_in,
                              void* d_out, int out_size, void* d_ws, size_t ws_size,
                              hipStream_t stream) {
  const float* qkv = (const float*)d_in[0];
  float* out = (float*)d_out;
  dim3 grid(kS / kBM, kH, kB);   // (32, 16, 4) workgroups
  fa_fwd_f16wmma<<<grid, 128, 0, stream>>>(qkv, out);
}
